// AttentionLayer_82471962017851
// MI455X (gfx1250) — compile-verified, matrix operand order checked by reading, not run on hardware
//
#include <hip/hip_runtime.h>

// ---------------------------------------------------------------------------
// Fused attention layer for MI455X (gfx1250, wave32, WMMA, async-LDS copies).
//   pass 0: x, wq, wk, wv, wf  -> bf16 in workspace
//   Q/K/V  = x@w^T + b          (bf16 WMMA GEMM, async double-buffered LDS)
//   per-token 8x8 head scores -> softmax -> context (VALU, tiny)
//   out    = context@wf^T + bf  (same GEMM, fp32 out)
// ---------------------------------------------------------------------------

typedef __attribute__((ext_vector_type(16))) __bf16 v16bf;
typedef __attribute__((ext_vector_type(8)))  float  v8f;

union Frag {
    v16bf v;
    uint4 q[2];
};

static constexpr int Hd  = 1024;   // hidden
static constexpr int NHd = 8;      // heads
static constexpr int HDd = 128;    // head dim
static constexpr int BK  = 32;     // K step (one bf16 WMMA K)
static constexpr int LDA = 40;     // LDS row stride in shorts (80B: 16B-aligned, bank-conflict-free)

__device__ __forceinline__ unsigned short f2bf(float f) {
    union { float f; unsigned u; } v; v.f = f;
    unsigned r = v.u + 0x7fffu + ((v.u >> 16) & 1u);   // round-to-nearest-even
    return (unsigned short)(r >> 16);
}
__device__ __forceinline__ unsigned pack2(float lo, float hi) {
    return (unsigned)f2bf(lo) | ((unsigned)f2bf(hi) << 16);
}
__device__ __forceinline__ float bflo(unsigned u) { return __uint_as_float(u << 16); }
__device__ __forceinline__ float bfhi(unsigned u) { return __uint_as_float(u & 0xffff0000u); }

#define WAIT_ASYNC4 asm volatile("s_wait_asynccnt 4" ::: "memory")
#define WAIT_ASYNC0 asm volatile("s_wait_asynccnt 0" ::: "memory")

// per-lane async copy: 16B global -> 16B LDS (ASYNCcnt-tracked, no VGPR data)
__device__ __forceinline__ void async_cp_b128(unsigned lds_addr, unsigned voff,
                                              const void* sbase, int ioff) {
    if (ioff == 0)
        asm volatile("global_load_async_to_lds_b128 %0, %1, %2 offset:0"
                     :: "v"(lds_addr), "v"(voff), "s"(sbase) : "memory");
    else
        asm volatile("global_load_async_to_lds_b128 %0, %1, %2 offset:16"
                     :: "v"(lds_addr), "v"(voff), "s"(sbase) : "memory");
}

// ---------------------------------------------------------------------------
// fp32 -> bf16 conversion (8 elements / thread)
// ---------------------------------------------------------------------------
__global__ __launch_bounds__(256)
void cvt_bf16(const float* __restrict__ src, unsigned short* __restrict__ dst, int n8) {
    const int i = blockIdx.x * blockDim.x + threadIdx.x;
    if (i >= n8) return;
    const float4 f0 = *(const float4*)(src + (size_t)i * 8);
    const float4 f1 = *(const float4*)(src + (size_t)i * 8 + 4);
    *(uint4*)(dst + (size_t)i * 8) = make_uint4(pack2(f0.x, f0.y), pack2(f0.z, f0.w),
                                                pack2(f1.x, f1.y), pack2(f1.z, f1.w));
}

// ---------------------------------------------------------------------------
// GEMM: out[M x 1024] = A_bf16[M x 1024] @ W_bf16[1024 x 1024]^T + bias
// 256 thr (8 waves), tile 128x128, wave = 2x4 WMMA subtiles,
// double-buffered LDS filled by GLOBAL_LOAD_ASYNC_TO_LDS_B128.
// ---------------------------------------------------------------------------
template <typename OT>
__global__ __launch_bounds__(256)
void gemm_bias_wmma(const unsigned short* __restrict__ A,
                    const unsigned short* __restrict__ W,
                    const float* __restrict__ bias,
                    OT* __restrict__ out) {
    constexpr int N = Hd, K = Hd;
    constexpr int PANEL = 128 * LDA;                 // ushorts per A (or B) panel
    __shared__ unsigned short Sh[2 * 2 * PANEL];     // [buf][A,B] = 40 KB

    const int t    = threadIdx.x;
    const int wave = t >> 5;
    const int lane = t & 31;
    const int wm   = wave & 3;                       // 4 waves along M (4*32=128)
    const int wn   = wave >> 2;                      // 2 waves along N (2*64=128)
    const int lrow = lane & 15;
    const int kbase = (lane >> 4) * 8;               // K sub-chunk per ISA A/B layout
    const int m0 = blockIdx.y * 128;
    const int n0 = blockIdx.x * 128;
    const int prow = t >> 1;                         // 0..127 panel row this thread stages
    const int half = t & 1;                          // which 16-element K half

    const unsigned shbase = (unsigned)(unsigned long long)(void*)Sh;  // LDS offset
    const unsigned ldsoff = (unsigned)((prow * LDA + half * 16) * 2);
    const unsigned aoff0  = ((unsigned)(m0 + prow) * (unsigned)K + (unsigned)(half * 16)) * 2u;
    const unsigned boff0  = ((unsigned)(n0 + prow) * (unsigned)K + (unsigned)(half * 16)) * 2u;
    constexpr unsigned BUFB = (unsigned)(2 * PANEL * 2);  // bytes per double-buffer slot

    auto issue = [&](int kstep, int buf) {
        const unsigned la = shbase + (unsigned)buf * BUFB + ldsoff;
        const unsigned lb = la + (unsigned)(PANEL * 2);
        const unsigned ga = aoff0 + (unsigned)kstep * (BK * 2u);
        const unsigned gb = boff0 + (unsigned)kstep * (BK * 2u);
        async_cp_b128(la, ga, A, 0);
        async_cp_b128(la, ga, A, 16);
        async_cp_b128(lb, gb, W, 0);
        async_cp_b128(lb, gb, W, 16);
    };

    v8f acc[2][4] = {};
    constexpr int nk = K / BK;                       // 32 K-steps

    issue(0, 0);
    for (int i = 0; i < nk; ++i) {
        const int buf = i & 1;
        if (i + 1 < nk) {                            // overlap next panel with compute
            issue(i + 1, buf ^ 1);
            WAIT_ASYNC4;                             // in-order: current panel landed
        } else {
            WAIT_ASYNC0;
        }
        __syncthreads();                             // all waves' panels visible

        const unsigned short* SA = Sh + buf * (2 * PANEL);
        const unsigned short* SB = SA + PANEL;

        Frag a[2], b[4];
#pragma unroll
        for (int ii = 0; ii < 2; ++ii) {
            const int r = wm * 32 + ii * 16 + lrow;
            a[ii].q[0] = *(const uint4*)&SA[r * LDA + kbase];
            a[ii].q[1] = *(const uint4*)&SA[r * LDA + kbase + 16];
        }
#pragma unroll
        for (int j = 0; j < 4; ++j) {
            const int n = wn * 64 + j * 16 + lrow;
            b[j].q[0] = *(const uint4*)&SB[n * LDA + kbase];
            b[j].q[1] = *(const uint4*)&SB[n * LDA + kbase + 16];
        }
#pragma unroll
        for (int ii = 0; ii < 2; ++ii)
#pragma unroll
            for (int j = 0; j < 4; ++j)
                acc[ii][j] = __builtin_amdgcn_wmma_f32_16x16x32_bf16(
                    false, a[ii].v, false, b[j].v, (short)0, acc[ii][j], false, false);

        __syncthreads();                             // reads done before next overwrite
    }

    // ---- epilogue: bias add + store (C/D layout: lane=N col, vgpr r=M row) ----
    const int col_l  = lane & 15;
    const int rowadd = (lane >> 4) * 8;
#pragma unroll
    for (int ii = 0; ii < 2; ++ii) {
        const int mbase = m0 + wm * 32 + ii * 16 + rowadd;
#pragma unroll
        for (int j = 0; j < 4; ++j) {
            const int col = n0 + wn * 64 + j * 16 + col_l;
            const float bv = bias[col];
#pragma unroll
            for (int r = 0; r < 8; ++r) {
                const float val = acc[ii][j][r] + bv;
                const size_t idx = (size_t)(mbase + r) * N + col;
                if constexpr (sizeof(OT) == 2) out[idx] = f2bf(val);
                else                           out[idx] = val;
            }
        }
    }
}

// ---------------------------------------------------------------------------
// Per-token head-vs-head attention: one thread per (token, q-head).
// ---------------------------------------------------------------------------
__global__ __launch_bounds__(256)
void attn_kernel(const unsigned short* __restrict__ Q,
                 const unsigned short* __restrict__ Kb,
                 const unsigned short* __restrict__ V,
                 float* __restrict__ attn_out,
                 unsigned short* __restrict__ ctx, int M) {
    const int t = blockIdx.x * blockDim.x + threadIdx.x;
    if (t >= M * NHd) return;
    const int pos = t >> 3;
    const int qh  = t & 7;
    const size_t base = (size_t)pos * Hd;

    uint4 qreg[16];
    const unsigned short* qp = Q + base + qh * HDd;
#pragma unroll
    for (int i = 0; i < 16; ++i) qreg[i] = *(const uint4*)(qp + i * 8);

    float sc[NHd];
#pragma unroll
    for (int kh = 0; kh < NHd; ++kh) {
        const unsigned short* kp = Kb + base + kh * HDd;
        float s = 0.f;
#pragma unroll
        for (int i = 0; i < 16; ++i) {
            const uint4 kq = *(const uint4*)(kp + i * 8);
            s += bflo(qreg[i].x) * bflo(kq.x) + bfhi(qreg[i].x) * bfhi(kq.x);
            s += bflo(qreg[i].y) * bflo(kq.y) + bfhi(qreg[i].y) * bfhi(kq.y);
            s += bflo(qreg[i].z) * bflo(kq.z) + bfhi(qreg[i].z) * bfhi(kq.z);
            s += bflo(qreg[i].w) * bflo(kq.w) + bfhi(qreg[i].w) * bfhi(kq.w);
        }
        sc[kh] = s * 0.08838834764831845f;   // 1/sqrt(128)
    }

    float mx = sc[0];
#pragma unroll
    for (int kh = 1; kh < NHd; ++kh) mx = fmaxf(mx, sc[kh]);
    float sum = 0.f;
#pragma unroll
    for (int kh = 0; kh < NHd; ++kh) { sc[kh] = __expf(sc[kh] - mx); sum += sc[kh]; }
    const float inv = 1.f / sum;
#pragma unroll
    for (int kh = 0; kh < NHd; ++kh) sc[kh] *= inv;

    float* ap = attn_out + (size_t)pos * (NHd * NHd) + qh * NHd;
    *(float4*)(ap + 0) = make_float4(sc[0], sc[1], sc[2], sc[3]);
    *(float4*)(ap + 4) = make_float4(sc[4], sc[5], sc[6], sc[7]);

    unsigned short* cp = ctx + base + qh * HDd;
#pragma unroll
    for (int dc = 0; dc < 16; ++dc) {
        float c[8] = {0.f, 0.f, 0.f, 0.f, 0.f, 0.f, 0.f, 0.f};
#pragma unroll
        for (int kh = 0; kh < NHd; ++kh) {
            const uint4 vq = *(const uint4*)(V + base + kh * HDd + dc * 8);
            const float p = sc[kh];
            c[0] += p * bflo(vq.x); c[1] += p * bfhi(vq.x);
            c[2] += p * bflo(vq.y); c[3] += p * bfhi(vq.y);
            c[4] += p * bflo(vq.z); c[5] += p * bfhi(vq.z);
            c[6] += p * bflo(vq.w); c[7] += p * bfhi(vq.w);
        }
        *(uint4*)(cp + dc * 8) = make_uint4(pack2(c[0], c[1]), pack2(c[2], c[3]),
                                            pack2(c[4], c[5]), pack2(c[6], c[7]));
    }
}

// ---------------------------------------------------------------------------
extern "C" void kernel_launch(void* const* d_in, const int* in_sizes, int n_in,
                              void* d_out, int out_size, void* d_ws, size_t ws_size,
                              hipStream_t stream) {
    const float* x   = (const float*)d_in[0];
    const float* wq  = (const float*)d_in[1];
    const float* bq  = (const float*)d_in[2];
    const float* wk  = (const float*)d_in[3];
    const float* bk  = (const float*)d_in[4];
    const float* wv  = (const float*)d_in[5];
    const float* bv  = (const float*)d_in[6];
    const float* wf  = (const float*)d_in[7];
    const float* bf_ = (const float*)d_in[8];

    const int M = in_sizes[0] / Hd;          // B*S = 32768
    const size_t MH = (size_t)M * Hd;
    const size_t HH = (size_t)Hd * Hd;

    // workspace (bf16): x, wq, wk, wv, wf, Q, K, V, context
    unsigned short* xb  = (unsigned short*)d_ws;
    unsigned short* wqb = xb  + MH;
    unsigned short* wkb = wqb + HH;
    unsigned short* wvb = wkb + HH;
    unsigned short* wfb = wvb + HH;
    unsigned short* Qb  = wfb + HH;
    unsigned short* Kbw = Qb  + MH;
    unsigned short* Vb  = Kbw + MH;
    unsigned short* Cb  = Vb  + MH;

    float* out  = (float*)d_out;             // (B,S,H) fp32
    float* attn = out + MH;                  // (B,S,nH,nH) fp32

    dim3 blk(256);

    // pass 0: convert operands to bf16
    cvt_bf16<<<(int)(MH / 8 / 256), blk, 0, stream>>>(x,  xb,  (int)(MH / 8));
    cvt_bf16<<<(int)(HH / 8 / 256), blk, 0, stream>>>(wq, wqb, (int)(HH / 8));
    cvt_bf16<<<(int)(HH / 8 / 256), blk, 0, stream>>>(wk, wkb, (int)(HH / 8));
    cvt_bf16<<<(int)(HH / 8 / 256), blk, 0, stream>>>(wv, wvb, (int)(HH / 8));
    cvt_bf16<<<(int)(HH / 8 / 256), blk, 0, stream>>>(wf, wfb, (int)(HH / 8));

    dim3 grd(Hd / 128, M / 128);             // (8, 256)
    gemm_bias_wmma<unsigned short><<<grd, blk, 0, stream>>>(xb, wqb, bq, Qb);
    gemm_bias_wmma<unsigned short><<<grd, blk, 0, stream>>>(xb, wkb, bk, Kbw);
    gemm_bias_wmma<unsigned short><<<grd, blk, 0, stream>>>(xb, wvb, bv, Vb);

    attn_kernel<<<(M * NHd + 255) / 256, blk, 0, stream>>>(Qb, Kbw, Vb, attn, Cb, M);

    gemm_bias_wmma<float><<<grd, blk, 0, stream>>>(Cb, wfb, bf_, out);
}
